// TransformerEncoderLayer_22995254903055
// MI455X (gfx1250) — matrix operand order repaired and checked by reading, not compiled, not run on hardware
//
#include <hip/hip_runtime.h>
#include <hip/hip_bf16.h>
#include <math.h>

// ---------------------------------------------------------------------------
// TransformerEncoderLayer for MI455X (gfx1250, wave32, WMMA bf16 16x16x32)
//   tokens N = 512*32 = 16384, D = 512, dff = 2048, B=32 H=8 S=512 dh=64
// ~200 GFLOP vs a few hundred MB HBM @ 23.3 TB/s -> matrix-op bound, so all
// contractions run through v_wmma_f32_16x16x32_bf16. Weights transposed to
// [N][K] bf16 once so B-fragments are contiguous 32B loads (L2-resident).
// GEMM A-tiles are double-buffered in LDS with async global->LDS loads
// (ASYNCcnt) overlapping the WMMA stream.
// ---------------------------------------------------------------------------

typedef __bf16 v16bf __attribute__((ext_vector_type(16)));
typedef __bf16 v8bf  __attribute__((ext_vector_type(8)));
typedef float  v8f   __attribute__((ext_vector_type(8)));

#define NTOK   16384
#define DMODEL 512
#define DFF    2048
#define L1B    32      // batch for attention
#define NHEAD  8
#define HDIM   64
#define SEQ    512

static __device__ inline v8f wmma_bf16(v16bf a, v16bf b, v8f c) {
    return __builtin_amdgcn_wmma_f32_16x16x32_bf16(
        false, a, false, b, (short)0, c, false, false);
}

// gfx1250 async global->LDS copy (16 bytes / lane), tracked by ASYNCcnt.
// VDST = 32-bit LDS byte address (generic LDS pointer low bits), VADDR = 64b.
// No "memory" clobber: all consumers sit behind __syncthreads(), and volatile
// asm statements keep order among themselves (issue always precedes wait).
static __device__ inline void async_ld16(void* lds, const void* g) {
    asm volatile("global_load_async_to_lds_b128 %0, %1, off"
                 :: "v"((unsigned int)(uintptr_t)lds), "v"(g));
}
static __device__ inline void wait_async0() {
    asm volatile("s_wait_asynccnt 0x0" ::: "memory");
}

static __device__ inline float wave_sum32(float v) {
#pragma unroll
    for (int m = 16; m >= 1; m >>= 1) v += __shfl_xor(v, m, 32);
    return v;
}

// ---------------------------------------------------------------------------
// f32 -> bf16 transposing weight conversion: in f32 [K][N] -> out bf16 [N][K]
// ---------------------------------------------------------------------------
__global__ __launch_bounds__(256) void cvt_transpose_kernel(
    const float* __restrict__ in, __bf16* __restrict__ out, int K, int N) {
    int idx = blockIdx.x * 256 + threadIdx.x;
    if (idx < K * N) {
        int k = idx / N, n = idx - k * N;
        out[(size_t)n * K + k] = (__bf16)in[idx];
    }
}

// ---------------------------------------------------------------------------
// Fused velocity / acceleration extraction + three layernorms (g1, beta1).
// One wave per token row; each lane owns 16 contiguous channels.
// ---------------------------------------------------------------------------
static __device__ inline void ln_store16(const float* vv, const float* __restrict__ g,
                                         const float* __restrict__ be, int lane,
                                         float* __restrict__ of, __bf16* __restrict__ ob) {
    float s = 0.f, s2 = 0.f;
#pragma unroll
    for (int e = 0; e < 16; e++) { s += vv[e]; s2 += vv[e] * vv[e]; }
    s  = wave_sum32(s);
    s2 = wave_sum32(s2);
    const float mu  = s * (1.f / (float)DMODEL);
    const float var = s2 * (1.f / (float)DMODEL) - mu * mu;
    const float rs  = rsqrtf(var + 1e-5f);
#pragma unroll
    for (int e = 0; e < 16; e++) {
        int c = lane * 16 + e;
        float o = (vv[e] - mu) * rs * g[c] + be[c];
        of[c] = o;
        ob[c] = (__bf16)o;
    }
}

__global__ __launch_bounds__(256) void vel_ln_kernel(
    const float* __restrict__ x, const float* __restrict__ g, const float* __restrict__ be,
    float* __restrict__ xnF, __bf16* __restrict__ xnB,
    float* __restrict__ vnF, __bf16* __restrict__ vnB,
    float* __restrict__ anF, __bf16* __restrict__ anB) {
    const int wid  = threadIdx.x >> 5;
    const int lane = threadIdx.x & 31;
    const int row  = blockIdx.x * 8 + wid;      // token index n = s*32 + b
    const int b    = row & (L1B - 1);
    const size_t base = (size_t)row * DMODEL + lane * 16;

    float x0[16], x1[16], x2[16], vl[16], ac[16];
#pragma unroll
    for (int e = 0; e < 16; e++) x0[e] = x[base + e];
#pragma unroll
    for (int e = 0; e < 16; e++) x1[e] = (b < 31) ? x[base + DMODEL + e] : 0.f;
#pragma unroll
    for (int e = 0; e < 16; e++) x2[e] = (b < 30) ? x[base + 2 * DMODEL + e] : 0.f;
#pragma unroll
    for (int e = 0; e < 16; e++) vl[e] = (b < 31) ? (x1[e] - x0[e]) : 0.f;
#pragma unroll
    for (int e = 0; e < 16; e++) {
        float a;
        if (b < 30)       a = x2[e] - 2.f * x1[e] + x0[e];
        else if (b == 30) a = x0[e] - x1[e];
        else              a = 0.f;
        ac[e] = a;
    }
    const size_t ro = (size_t)row * DMODEL;
    ln_store16(x0, g, be, lane, xnF + ro, xnB + ro);
    ln_store16(vl, g, be, lane, vnF + ro, vnB + ro);
    ln_store16(ac, g, be, lane, anF + ro, anB + ro);
}

__global__ __launch_bounds__(256) void ln2_kernel(
    const float* __restrict__ S, const float* __restrict__ g, const float* __restrict__ be,
    float* __restrict__ hF, __bf16* __restrict__ hB) {
    const int wid  = threadIdx.x >> 5;
    const int lane = threadIdx.x & 31;
    const int row  = blockIdx.x * 8 + wid;
    const size_t base = (size_t)row * DMODEL + lane * 16;
    float v[16];
#pragma unroll
    for (int e = 0; e < 16; e++) v[e] = S[base + e];
    const size_t ro = (size_t)row * DMODEL;
    ln_store16(v, g, be, lane, hF + ro, hB + ro);
}

// ---------------------------------------------------------------------------
// bf16 WMMA GEMM:  out = epi(A[M,K] @ B^T[N,K]^T + bias)
// Block tile 128x128, 8 waves (4M x 2N), wave tile 32x64 (2x4 WMMA tiles).
// A double-buffered in LDS: async global->LDS copy of tile k+1 overlaps the
// WMMA stream on tile k (canonical ASYNCcnt producer/consumer pattern).
// B-frags read straight from the transposed L2-resident weight.
// Epilogue fully templated: no per-element branches.
// ---------------------------------------------------------------------------
static __device__ inline v16bf lds_a_frag(const __bf16* At, int rowBase, int lane) {
    const int row = rowBase + (lane & 15);
    const int k0  = (lane < 16) ? 0 : 16;
    const __bf16* p = At + row * 32 + k0;
    v8bf lo = *(const v8bf*)p;
    v8bf hi = *(const v8bf*)(p + 8);
    v16bf r;
#pragma unroll
    for (int i = 0; i < 8; i++) { r[i] = lo[i]; r[i + 8] = hi[i]; }
    return r;
}

template <int RELU, int HAS_RES, int ACCUM, int WF32, int WBF16>
__global__ __launch_bounds__(256) void gemm_bf16_kernel(
    const __bf16* __restrict__ A, const __bf16* __restrict__ Bt,  // Bt: [N][K]
    const float* __restrict__ bias, const float* __restrict__ res,
    float* __restrict__ outF, __bf16* __restrict__ outB,
    int M, int N, int K) {
    __shared__ alignas(64) __bf16 At0[128 * 32];
    __shared__ alignas(64) __bf16 At1[128 * 32];

    const int tid  = threadIdx.x;
    const int lane = tid & 31;
    const int wid  = tid >> 5;
    const int wm   = wid & 3;         // M offset wm*32
    const int wn   = wid >> 2;        // N offset wn*64
    const int bm   = blockIdx.x * 128;
    const int bn   = blockIdx.y * 128;

    v8f acc[2][4];
#pragma unroll
    for (int tm = 0; tm < 2; tm++)
#pragma unroll
        for (int tn = 0; tn < 4; tn++) acc[tm][tn] = (v8f){0.f,0.f,0.f,0.f,0.f,0.f,0.f,0.f};

    const int aRow = tid >> 1, aCol = (tid & 1) * 16;
    const __bf16* aSrc = A + (size_t)(bm + aRow) * K + aCol;
    __bf16* aDst0 = At0 + aRow * 32 + aCol;
    __bf16* aDst1 = At1 + aRow * 32 + aCol;

    // prologue: stage tile 0 into buffer 0
    async_ld16(aDst0,     aSrc);
    async_ld16(aDst0 + 8, aSrc + 8);
    wait_async0();
    __syncthreads();

    for (int kk = 0; kk < K; kk += 32) {
        const bool pong = (kk >> 5) & 1;
        const __bf16* curT = pong ? At1 : At0;
        __bf16* nxtD = pong ? aDst0 : aDst1;

        // issue async copy of tile k+1 into the other buffer (overlaps WMMA)
        if (kk + 32 < K) {
            async_ld16(nxtD,     aSrc + kk + 32);
            async_ld16(nxtD + 8, aSrc + kk + 40);
            __builtin_prefetch(aSrc + kk + 64, 0, 3);   // global_prefetch_b8
        }

        v16bf af[2];
#pragma unroll
        for (int tm = 0; tm < 2; tm++) af[tm] = lds_a_frag(curT, wm * 32 + tm * 16, lane);
#pragma unroll
        for (int tn = 0; tn < 4; tn++) {
            const int col = bn + wn * 64 + tn * 16 + (lane & 15);
            const int k0  = kk + ((lane < 16) ? 0 : 16);
            v16bf bfg = *(const v16bf*)(Bt + (size_t)col * K + k0);
#pragma unroll
            for (int tm = 0; tm < 2; tm++)
                acc[tm][tn] = wmma_bf16(af[tm], bfg, acc[tm][tn]);
        }

        wait_async0();
        __syncthreads();
    }

    // epilogue; C layout: vgpr i -> M = i + 8*(lane>=16), N = lane%16
#pragma unroll
    for (int tm = 0; tm < 2; tm++) {
#pragma unroll
        for (int tn = 0; tn < 4; tn++) {
            const int rowBase = bm + wm * 32 + tm * 16 + 8 * (lane >> 4);
            const int col     = bn + wn * 64 + tn * 16 + (lane & 15);
            const float bc    = bias[col];
            size_t idx = (size_t)rowBase * N + col;
#pragma unroll
            for (int i = 0; i < 8; i++, idx += N) {
                float v = acc[tm][tn][i] + bc;
                if (RELU)    v = fmaxf(v, 0.f);
                if (HAS_RES) v += res[idx];
                if (ACCUM)   v += outF[idx];
                if (WF32)    outF[idx] = v;
                if (WBF16)   outB[idx] = (__bf16)v;
            }
        }
    }
}

// ---------------------------------------------------------------------------
// Attention for one (batch b, head h, 64-query tile).
// Faithful quirk: scores /= HDIM (not sqrt(HDIM)).
// LDS (256 KB of the 320 KB WGP budget):
//   sc : 64x512 f32 scores                (128 KB)
//   Pb : 64x512 bf16 normalized probs     ( 64 KB)
//   Kb : 512x64 bf16 K/V head panel       ( 64 KB, async-staged)
// Phase 0: async-stage K/V panel.  Phase A: S = Q K^T (WMMA).
// Phase B: softmax.  Phase C: O = P V (WMMA; P and V both from LDS).
// ---------------------------------------------------------------------------
__global__ __launch_bounds__(256) void attn_kernel(
    const __bf16* __restrict__ Q, const __bf16* __restrict__ KV,
    __bf16* __restrict__ O) {
    extern __shared__ __align__(64) char smem[];
    float*  sc = (float*)smem;                               // [64][512]
    __bf16* Pb = (__bf16*)(smem + 64 * SEQ * 4);             // [64][512]
    __bf16* Kb = (__bf16*)(smem + 64 * SEQ * 4 + 64 * SEQ * 2); // [512][64]

    const int tid  = threadIdx.x;
    const int lane = tid & 31;
    const int wid  = tid >> 5;
    const int bidx = blockIdx.x;
    const int qt   = bidx & 7;                    // 64-query tile
    const int h    = (bidx >> 3) & 7;
    const int b    = bidx >> 6;
    const int qbase = qt * 64;
    const size_t hoff = (size_t)h * HDIM;

    auto tok = [&](int s) -> size_t { return ((size_t)(s * L1B + b)) * DMODEL + hoff; };

    // ---- Phase 0: async-stage the 512x64 K/V head panel into LDS ----
#pragma unroll
    for (int r = 0; r < 2; r++) {
        const int key = tid * 2 + r;
        const __bf16* src = KV + tok(key);
        __bf16* dst = Kb + key * HDIM;
#pragma unroll
        for (int c = 0; c < 8; c++)
            async_ld16(dst + c * 8, src + c * 8);
    }
    wait_async0();
    __syncthreads();

    // ---- Phase A: scores = Q K^T ----
    const int kcbase = wid * 64;                  // this wave's 64 keys
    for (int tq = 0; tq < 4; tq++) {
        v16bf aq[2];
#pragma unroll
        for (int t = 0; t < 2; t++) {
            const int qs = qbase + tq * 16 + (lane & 15);
            const int d0 = 32 * t + ((lane < 16) ? 0 : 16);
            const __bf16* p = Q + tok(qs) + d0;
            v8bf lo = *(const v8bf*)p;
            v8bf hi = *(const v8bf*)(p + 8);
#pragma unroll
            for (int i = 0; i < 8; i++) { aq[t][i] = lo[i]; aq[t][i + 8] = hi[i]; }
        }
#pragma unroll
        for (int tk = 0; tk < 4; tk++) {
            v8f accv = (v8f){0.f,0.f,0.f,0.f,0.f,0.f,0.f,0.f};
#pragma unroll
            for (int t = 0; t < 2; t++) {
                const int key = kcbase + tk * 16 + (lane & 15);
                const int d0  = 32 * t + ((lane < 16) ? 0 : 16);
                v16bf bv = *(const v16bf*)(Kb + key * HDIM + d0);
                accv = wmma_bf16(aq[t], bv, accv);
            }
            const int rr = tq * 16 + 8 * (lane >> 4);
            const int cc = kcbase + tk * 16 + (lane & 15);
#pragma unroll
            for (int i = 0; i < 8; i++) sc[(rr + i) * SEQ + cc] = accv[i];
        }
    }
    __syncthreads();

    // ---- Phase B: softmax(scores / 64) per row; emit bf16 probs ----
    if (tid < 64) {
        float* rowp = sc + tid * SEQ;
        __bf16* prow = Pb + tid * SEQ;
        float mx = -INFINITY;
        for (int j = 0; j < SEQ; j++) mx = fmaxf(mx, rowp[j]);
        float s = 0.f;
        const float inv = 1.0f / (float)HDIM;
        for (int j = 0; j < SEQ; j++) {
            float e = __expf((rowp[j] - mx) * inv);
            rowp[j] = e;
            s += e;
        }
        const float rinv = 1.0f / s;
        for (int j = 0; j < SEQ; j++) prow[j] = (__bf16)(rowp[j] * rinv);
    }
    __syncthreads();

    // ---- Phase C: O = P V ----
    const int tq  = wid >> 1;                     // row group (16 rows)
    const int dhb = (wid & 1) * 32;               // dh half
#pragma unroll
    for (int tn = 0; tn < 2; tn++) {
        v8f accv = (v8f){0.f,0.f,0.f,0.f,0.f,0.f,0.f,0.f};
        for (int kk = 0; kk < SEQ / 32; kk++) {
            // A-frag: probabilities (bf16, contiguous in LDS)
            const int prow = tq * 16 + (lane & 15);
            const int k0   = 32 * kk + ((lane < 16) ? 0 : 16);
            const __bf16* pp = Pb + prow * SEQ + k0;
            v8bf lo = *(const v8bf*)pp;
            v8bf hi = *(const v8bf*)(pp + 8);
            v16bf ap;
#pragma unroll
            for (int i = 0; i < 8; i++) { ap[i] = lo[i]; ap[i + 8] = hi[i]; }

            // B-frag: V columns gathered from LDS panel
            v16bf bv;
            const int col = dhb + tn * 16 + (lane & 15);
            const int kb  = 32 * kk + ((lane < 16) ? 0 : 16);
#pragma unroll
            for (int e = 0; e < 16; e++) bv[e] = Kb[(kb + e) * HDIM + col];
            accv = wmma_bf16(ap, bv, accv);
        }
        const int rr = qbase + tq * 16 + 8 * (lane >> 4);
        const int cc = dhb + tn * 16 + (lane & 15);
#pragma unroll
        for (int i = 0; i < 8; i++) O[tok(rr + i) + cc] = (__bf16)accv[i];
    }
}

// ---------------------------------------------------------------------------
// Host-side orchestration
// ---------------------------------------------------------------------------
extern "C" void kernel_launch(void* const* d_in, const int* in_sizes, int n_in,
                              void* d_out, int out_size, void* d_ws, size_t ws_size,
                              hipStream_t stream) {
    (void)in_sizes; (void)n_in; (void)out_size; (void)ws_size;

    const float* x     = (const float*)d_in[0];
    const float* Wq    = (const float*)d_in[1];
    const float* bq    = (const float*)d_in[2];
    const float* Wo    = (const float*)d_in[3];
    const float* bo    = (const float*)d_in[4];
    const float* W1    = (const float*)d_in[5];
    const float* b1    = (const float*)d_in[6];
    const float* W2    = (const float*)d_in[7];
    const float* b2    = (const float*)d_in[8];
    const float* g1    = (const float*)d_in[9];
    const float* beta1 = (const float*)d_in[10];
    const float* g2    = (const float*)d_in[11];
    const float* beta2 = (const float*)d_in[12];
    float* out = (float*)d_out;

    char* cur = (char*)d_ws;
    auto alloc = [&](size_t bytes) -> void* {
        uintptr_t p = (uintptr_t)cur;
        p = (p + 255) & ~(uintptr_t)255;
        cur = (char*)(p + bytes);
        return (void*)p;
    };
    const size_t NT  = (size_t)NTOK * DMODEL;
    const size_t NTF = (size_t)NTOK * DFF;

    float*  xnF  = (float*) alloc(NT * 4);
    float*  vnF  = (float*) alloc(NT * 4);
    float*  anF  = (float*) alloc(NT * 4);
    float*  Ssum = (float*) alloc(NT * 4);
    float*  hF   = (float*) alloc(NT * 4);
    __bf16* xnB  = (__bf16*)alloc(NT * 2);
    __bf16* vnB  = (__bf16*)alloc(NT * 2);
    __bf16* anB  = (__bf16*)alloc(NT * 2);
    __bf16* P1   = (__bf16*)alloc(NT * 2);   // qkv projection for MHA1
    __bf16* Obuf = (__bf16*)alloc(NT * 2);   // attention output (reused x3)
    __bf16* xpB  = (__bf16*)alloc(NT * 2);   // bf16(x_p)
    __bf16* Q2   = (__bf16*)alloc(NT * 2);   // shared query proj for MHA2/3
    __bf16* KVb  = (__bf16*)alloc(NT * 2);   // key/value proj (reused)
    __bf16* hB   = (__bf16*)alloc(NT * 2);
    __bf16* ffn1 = (__bf16*)alloc(NTF * 2);
    __bf16* WqT  = (__bf16*)alloc((size_t)DMODEL * DMODEL * 2);  // [N][K]
    __bf16* WoT  = (__bf16*)alloc((size_t)DMODEL * DMODEL * 2);
    __bf16* W1T  = (__bf16*)alloc((size_t)DMODEL * DFF * 2);
    __bf16* W2T  = (__bf16*)alloc((size_t)DFF * DMODEL * 2);

    // --- weights to bf16, transposed to [N][K] ---
    const int wsz = DMODEL * DMODEL, fsz = DMODEL * DFF;
    cvt_transpose_kernel<<<(wsz + 255) / 256, 256, 0, stream>>>(Wq, WqT, DMODEL, DMODEL);
    cvt_transpose_kernel<<<(wsz + 255) / 256, 256, 0, stream>>>(Wo, WoT, DMODEL, DMODEL);
    cvt_transpose_kernel<<<(fsz + 255) / 256, 256, 0, stream>>>(W1, W1T, DMODEL, DFF);
    cvt_transpose_kernel<<<(fsz + 255) / 256, 256, 0, stream>>>(W2, W2T, DFF, DMODEL);

    // --- velocity / acceleration + layernorms (g1) ---
    vel_ln_kernel<<<NTOK / 8, 256, 0, stream>>>(x, g1, beta1, xnF, xnB, vnF, vnB, anF, anB);

    const dim3 gProj(NTOK / 128, DMODEL / 128);        // 128 x 4
    const dim3 gFF1 (NTOK / 128, DFF / 128);           // 128 x 16
    const int  attnGrid = L1B * NHEAD * (SEQ / 64);    // 2048
    const size_t attnLds = (size_t)64 * SEQ * 4 + (size_t)64 * SEQ * 2
                         + (size_t)SEQ * HDIM * 2;     // 256 KB

    // ===== MHA1 (self): q=k=v = xn @ Wq + bq =====
    gemm_bf16_kernel<0,0,0,0,1><<<gProj, 256, 0, stream>>>(
        xnB, WqT, bq, nullptr, nullptr, P1, NTOK, DMODEL, DMODEL);
    attn_kernel<<<attnGrid, 256, attnLds, stream>>>(P1, P1, Obuf);
    // S = xn + O1 @ Wo + bo ; xpB = bf16(S)
    gemm_bf16_kernel<0,1,0,1,1><<<gProj, 256, 0, stream>>>(
        Obuf, WoT, bo, xnF, Ssum, xpB, NTOK, DMODEL, DMODEL);

    // ===== shared query projection for MHA2/MHA3: Q2 = x_p @ Wq + bq =====
    gemm_bf16_kernel<0,0,0,0,1><<<gProj, 256, 0, stream>>>(
        xpB, WqT, bq, nullptr, nullptr, Q2, NTOK, DMODEL, DMODEL);

    // ===== MHA2 (cross with velocity): kv = vel_n @ Wq + bq =====
    gemm_bf16_kernel<0,0,0,0,1><<<gProj, 256, 0, stream>>>(
        vnB, WqT, bq, nullptr, nullptr, KVb, NTOK, DMODEL, DMODEL);
    attn_kernel<<<attnGrid, 256, attnLds, stream>>>(Q2, KVb, Obuf);
    // S += vel_n + O2 @ Wo + bo
    gemm_bf16_kernel<0,1,1,1,0><<<gProj, 256, 0, stream>>>(
        Obuf, WoT, bo, vnF, Ssum, nullptr, NTOK, DMODEL, DMODEL);

    // ===== MHA3 (cross with acceleration): kv = acc_n @ Wq + bq =====
    gemm_bf16_kernel<0,0,0,0,1><<<gProj, 256, 0, stream>>>(
        anB, WqT, bq, nullptr, nullptr, KVb, NTOK, DMODEL, DMODEL);
    attn_kernel<<<attnGrid, 256, attnLds, stream>>>(Q2, KVb, Obuf);
    // S += acc_n + O3 @ Wo + bo
    gemm_bf16_kernel<0,1,1,1,0><<<gProj, 256, 0, stream>>>(
        Obuf, WoT, bo, anF, Ssum, nullptr, NTOK, DMODEL, DMODEL);

    // ===== h = LN(S, g2, beta2) =====
    ln2_kernel<<<NTOK / 8, 256, 0, stream>>>(Ssum, g2, beta2, hF, hB);

    // ===== FFN =====
    gemm_bf16_kernel<1,0,0,0,1><<<gFF1, 256, 0, stream>>>(
        hB, W1T, b1, nullptr, nullptr, ffn1, NTOK, DFF, DMODEL);
    gemm_bf16_kernel<0,1,0,1,0><<<gProj, 256, 0, stream>>>(
        ffn1, W2T, b2, hF, out, nullptr, NTOK, DMODEL, DFF);
}
